// PositionalAttention_4312147165581
// MI455X (gfx1250) — matrix operand order repaired and checked by reading, not compiled, not run on hardware
//
#include <hip/hip_runtime.h>

typedef __attribute__((ext_vector_type(16))) _Float16 v16h;
typedef __attribute__((ext_vector_type(8)))  _Float16 v8h;
typedef __attribute__((ext_vector_type(8)))  float    v8f;

#define CAT16(a,b) __builtin_shufflevector((a),(b),0,1,2,3,4,5,6,7,8,9,10,11,12,13,14,15)
#define WMMA(a,b,c) __builtin_amdgcn_wmma_f32_16x16x32_f16(false,(a),false,(b),(short)0,(c),false,false)

static constexpr int BB = 4, CC = 256, NN = 4096, DD = 32;

// lane <-> lane^16 exchange (group-of-32 swizzle, xor_mask=0x10, and_mask=0x1f)
__device__ __forceinline__ float swz16(float x) {
  return __int_as_float(__builtin_amdgcn_ds_swizzle(__float_as_int(x), 0x401F));
}

// CDNA5 async global->LDS copies (tracked by ASYNCcnt)
__device__ __forceinline__ void async_ld_b128(unsigned lds, const void* gaddr) {
  asm volatile("global_load_async_to_lds_b128 %0, %1, off" :: "v"(lds), "v"(gaddr) : "memory");
}
__device__ __forceinline__ void async_ld_b64(unsigned lds, const void* gaddr) {
  asm volatile("global_load_async_to_lds_b64 %0, %1, off" :: "v"(lds), "v"(gaddr) : "memory");
}
__device__ __forceinline__ void wait_async0() {
  asm volatile("s_wait_asynccnt 0x0" ::: "memory");
}

// ---------------------------------------------------------------- prep: weights -> f16 concat [320][256], bias concat [320]
__global__ void prep_weights(const float* __restrict__ Wq, const float* __restrict__ bq,
                             const float* __restrict__ Wk, const float* __restrict__ bk,
                             const float* __restrict__ Wv, const float* __restrict__ bv,
                             _Float16* __restrict__ Wh, float* __restrict__ biasc) {
  int idx = blockIdx.x * 256 + threadIdx.x;
  if (idx < 320 * 256) {
    int row = idx >> 8, col = idx & 255;
    float w;
    if (row < 32)      w = Wq[row * 256 + col];
    else if (row < 64) w = Wk[(row - 32) * 256 + col];
    else               w = Wv[(row - 64) * 256 + col];
    Wh[idx] = (_Float16)w;
  }
  if (idx < 320) {
    float b;
    if (idx < 32)      b = bq[idx];
    else if (idx < 64) b = bk[idx - 32];
    else               b = bv[idx - 64];
    biasc[idx] = b;
  }
}

// ---------------------------------------------------------------- prep: x [b][c][n] f32 -> xT [b][n][c] f16
__global__ void prep_x(const float* __restrict__ x, _Float16* __restrict__ xT) {
  int idx = blockIdx.x * 256 + threadIdx.x;       // B*N*C = 2^22
  int c = idx & 255;
  int n = (idx >> 8) & (NN - 1);
  int b = idx >> 20;
  xT[idx] = (_Float16)x[((size_t)(b * CC + c)) * NN + n];
}

// ---------------------------------------------------------------- projections: [n][cout] = xT[n][:] . Wh[cout][:] + bias
// one wave per 16x16 tile; K=256 in 8 wmma steps
__global__ void __launch_bounds__(256) proj_gemm(const _Float16* __restrict__ xT,
                                                 const _Float16* __restrict__ Wh,
                                                 const float* __restrict__ biasc,
                                                 _Float16* __restrict__ qT,
                                                 _Float16* __restrict__ kT,
                                                 _Float16* __restrict__ vT) {
  int wid = threadIdx.x >> 5, lane = threadIdx.x & 31;
  int hi = lane >> 4, ln = lane & 15;
  int tile = blockIdx.x * 8 + wid;                 // 0..20479
  int ct = tile % 20;
  int nt = (tile / 20) & 255;
  int b  = tile / (20 * 256);
  int n0 = nt * 16;
  int cout = ct * 16 + ln;

  v8f acc;
  float bias = biasc[cout];
#pragma unroll
  for (int r = 0; r < 8; r++) acc[r] = bias;

  const _Float16* arow = xT + ((size_t)(b * NN + n0 + ln)) * CC;   // A: row n = lane%16
  const _Float16* brow = Wh + (size_t)cout * CC;                   // B: col cout = lane%16
#pragma unroll
  for (int ks = 0; ks < 8; ks++) {
    int cb = ks * 32;
    v8h a0 = *(const v8h*)(arow + cb + hi * 8);        // K = hi*8 .. +7
    v8h a1 = *(const v8h*)(arow + cb + 16 + hi * 8);   // K = 16+hi*8 .. +7
    v16h A = CAT16(a0, a1);
    v8h b0 = *(const v8h*)(brow + cb + hi * 8);        // K = hi*8 .. +7
    v8h b1 = *(const v8h*)(brow + cb + 16 + hi * 8);
    v16h Bv = CAT16(b0, b1);
    acc = WMMA(A, Bv, acc);
  }

  // C layout: VGPR r -> row n = n0 + r + 8*hi ; lane%16 -> cout
  if (ct < 4) {
    _Float16* dst = (ct < 2) ? qT : kT;
    int d = (ct & 1) * 16 + ln;
#pragma unroll
    for (int r = 0; r < 8; r++) {
      int n = n0 + r + hi * 8;
      dst[((size_t)(b * NN + n)) * DD + d] = (_Float16)acc[r];
    }
  } else {
    int c = (ct - 4) * 16 + ln;
    v8h pk;
#pragma unroll
    for (int r = 0; r < 8; r++) pk[r] = (_Float16)acc[r];
    *(v8h*)(vT + ((size_t)(b * CC + c)) * NN + n0 + hi * 8) = pk;
  }
}

// ---------------------------------------------------------------- fused flash attention
// one wave = 16 queries. S^T = Ktile x q  (C-layout: lane%16 = i, VGPR r + 8*hi = j)
// O^T = Vtile x P^T accumulated over 16 c-tiles. 8 waves/block share double-buffered
// K/V LDS chunks staged with CDNA5 async global->LDS loads.
__global__ void __launch_bounds__(256) attn(const _Float16* __restrict__ qT,
                                            const _Float16* __restrict__ kT,
                                            const _Float16* __restrict__ vT,
                                            const float* __restrict__ x,
                                            const float* __restrict__ gamma_p,
                                            float* __restrict__ out) {
  // per buffer: K tile [32 j][32 d] = 2048 B, V tile [256 c][32 j] = 16384 B
  __shared__ __align__(16) unsigned char smem[2 * 18432];

  int tid = threadIdx.x;
  int wid = tid >> 5, lane = tid & 31, hi = lane >> 4, ln = lane & 15;
  int b  = blockIdx.x >> 5;
  int i0 = ((blockIdx.x & 31) * 8 + wid) * 16;

  const _Float16* kbase = kT + (size_t)b * NN * DD;
  const _Float16* vbase = vT + (size_t)b * CC * NN;
  unsigned smem_lds = (unsigned)(uintptr_t)(&smem[0]);   // LDS byte offset = addr[31:0]

  // issue async copies of one 32-key chunk into buffer `buf` (5 uniform ops/thread)
  auto issue = [&](int j0, int buf) {
    unsigned base = smem_lds + buf * 18432;
    // K: contiguous 2 KB, thread tid copies 8 B
    async_ld_b64(base + tid * 8, (const char*)kbase + (size_t)j0 * DD * 2 + tid * 8);
    // V: thread tid = row c, 64 B
    const char* vsrc = (const char*)(vbase + (size_t)tid * NN + j0);
    unsigned vdst = base + 2048 + tid * 64;
    async_ld_b128(vdst,      vsrc);
    async_ld_b128(vdst + 16, vsrc + 16);
    async_ld_b128(vdst + 32, vsrc + 32);
    async_ld_b128(vdst + 48, vsrc + 48);
  };

  // q as B operand: lane%16 = i, K(d) = hi*8..+7 then 16+hi*8..+7
  const _Float16* qrow = qT + ((size_t)(b * NN + i0 + ln)) * DD + hi * 8;
  v8h q0 = *(const v8h*)qrow;
  v8h q1 = *(const v8h*)(qrow + 16);
  v16h Bq = CAT16(q0, q1);

  v8f O[16];
#pragma unroll
  for (int t = 0; t < 16; t++)
#pragma unroll
    for (int r = 0; r < 8; r++) O[t][r] = 0.0f;

  float mrun = -3.0e38f, lrun = 0.0f;

  issue(0, 0);   // prologue: chunk 0 -> buffer 0

  for (int jc = 0; jc < NN / 32; jc++) {
    int cur = jc & 1;
    wait_async0();        // current chunk landed in LDS
    __syncthreads();      // all waves: data visible AND previous compute (other buffer) done
    if (jc + 1 < NN / 32) issue((jc + 1) * 32, cur ^ 1);   // prefetch next chunk

    const _Float16* sK = (const _Float16*)(smem + cur * 18432);
    const _Float16* sV = (const _Float16*)(smem + cur * 18432 + 2048);

    // S^T tiles: A = K rows (j), B = q
    v8h ka = *(const v8h*)(sK + ln * 32 + hi * 8);
    v8h kb = *(const v8h*)(sK + ln * 32 + 16 + hi * 8);
    v16h A1 = CAT16(ka, kb);
    v8h kc = *(const v8h*)(sK + (16 + ln) * 32 + hi * 8);
    v8h kd = *(const v8h*)(sK + (16 + ln) * 32 + 16 + hi * 8);
    v16h A2 = CAT16(kc, kd);

    v8f S1, S2;
#pragma unroll
    for (int r = 0; r < 8; r++) { S1[r] = 0.0f; S2[r] = 0.0f; }
    S1 = WMMA(A1, Bq, S1);   // rows j0 + r + 8*hi
    S2 = WMMA(A2, Bq, S2);   // rows j0+16 + r + 8*hi

    // online softmax over j (in-lane over 8 VGPRs x 2 tiles, one xor-16 for cross-half)
    float mloc = -3.0e38f;
#pragma unroll
    for (int r = 0; r < 8; r++) mloc = fmaxf(mloc, fmaxf(S1[r], S2[r]));
    float mnew = fmaxf(mrun, fmaxf(mloc, swz16(mloc)));
    float corr = __expf(mrun - mnew);
    float p1[8], p2[8], ssum = 0.0f;
#pragma unroll
    for (int r = 0; r < 8; r++) {
      p1[r] = __expf(S1[r] - mnew);
      p2[r] = __expf(S2[r] - mnew);
      ssum += p1[r] + p2[r];
    }
    lrun = lrun * corr + ssum + swz16(ssum);
    mrun = mnew;

    // P^T as B operand: lo lanes need j 0..15 (own P1 + partner P1), hi need 16..31
    v16h Pb;
#pragma unroll
    for (int r = 0; r < 8; r++) {
      float o1 = swz16(p1[r]);
      float o2 = swz16(p2[r]);
      Pb[r]     = (_Float16)(hi ? o2 : p1[r]);
      Pb[r + 8] = (_Float16)(hi ? p2[r] : o1);
    }

    // O^T[ct] += V[ct] x P^T  (rescale by corr folded in)
#pragma unroll
    for (int ct = 0; ct < 16; ct++) {
      int c = ct * 16 + ln;
      v8h va = *(const v8h*)(sV + c * 32 + hi * 16);
      v8h vb = *(const v8h*)(sV + c * 32 + hi * 16 + 8);
      v16h Av = CAT16(va, vb);
      v8f o = O[ct];
#pragma unroll
      for (int r = 0; r < 8; r++) o[r] *= corr;
      O[ct] = WMMA(Av, Pb, o);
    }
  }

  float inv = 1.0f / lrun;
  float g = gamma_p[0];
#pragma unroll
  for (int ct = 0; ct < 16; ct++) {
#pragma unroll
    for (int r = 0; r < 8; r++) {
      int c = ct * 16 + r + hi * 8;
      size_t idx = ((size_t)(b * CC + c)) * NN + i0 + ln;
      out[idx] = g * O[ct][r] * inv + x[idx];
    }
  }
}

// ---------------------------------------------------------------- launch
extern "C" void kernel_launch(void* const* d_in, const int* in_sizes, int n_in,
                              void* d_out, int out_size, void* d_ws, size_t ws_size,
                              hipStream_t stream) {
  const float* x  = (const float*)d_in[0];
  const float* Wq = (const float*)d_in[1];
  const float* bq = (const float*)d_in[2];
  const float* Wk = (const float*)d_in[3];
  const float* bk = (const float*)d_in[4];
  const float* Wv = (const float*)d_in[5];
  const float* bv = (const float*)d_in[6];
  const float* gm = (const float*)d_in[7];
  float* out = (float*)d_out;

  char* ws = (char*)d_ws;
  _Float16* Wh    = (_Float16*)(ws);                       // 320*256*2   = 163840
  float*    biasc = (float*)(ws + 163840);                 // 320*4
  _Float16* xT    = (_Float16*)(ws + 262144);              // 4*4096*256*2 = 8 MiB
  _Float16* qT    = (_Float16*)(ws + 262144 + 8388608);    // 1 MiB
  _Float16* kT    = qT + (size_t)BB * NN * DD;             // 1 MiB
  _Float16* vT    = kT + (size_t)BB * NN * DD;             // 8 MiB   (total ~18.3 MiB)

  prep_weights<<<320, 256, 0, stream>>>(Wq, bq, Wk, bk, Wv, bv, Wh, biasc);
  prep_x<<<(BB * NN * CC) / 256, 256, 0, stream>>>(x, xT);
  proj_gemm<<<2560, 256, 0, stream>>>(xT, Wh, biasc, qT, kT, vT);
  attn<<<128, 256, 0, stream>>>(qT, kT, vT, x, gm, out);
}
